// TopKSparseTransformerLayer_32547262169107
// MI455X (gfx1250) — matrix operand order repaired and checked by reading, not compiled, not run on hardware
//
#include <hip/hip_runtime.h>
#include <hip/hip_bf16.h>
#include <math.h>

// ---------------- types ----------------
typedef unsigned short u16;
typedef __attribute__((ext_vector_type(16))) __bf16           v16bf;
typedef __attribute__((ext_vector_type(8)))  float            v8f;
typedef __attribute__((ext_vector_type(16))) unsigned short   v16u;
typedef __attribute__((ext_vector_type(8)))  unsigned short   v8u;
typedef __attribute__((ext_vector_type(4)))  unsigned         v4ui;
typedef __attribute__((ext_vector_type(4)))  int              v4i;
typedef __attribute__((ext_vector_type(8)))  int              v8i;

#define EMBED  1024
#define NHEAD  16
#define HDIM   64
#define DFF    4096
#define SEQ    2048
#define BATCH  2
#define TOK    (BATCH*SEQ)   // 4096
#define TOPK   64

// ---- gfx1250 feature probes (device pass only; host pass parses fallbacks) ----
#if defined(__has_builtin)
#if __has_builtin(__builtin_amdgcn_global_load_async_to_lds_b32)
#define HAVE_ASYNC_COPY 1
#endif
#if __has_builtin(__builtin_amdgcn_tensor_load_to_lds) && __has_builtin(__builtin_amdgcn_s_wait_tensorcnt)
#define HAVE_TDM 1
#endif
#endif

__device__ __forceinline__ u16 f2bf(float f) {
  union { float f; unsigned u; } v; v.f = f;
  unsigned r = v.u + 0x7FFFu + ((v.u >> 16) & 1u);   // round-to-nearest-even
  return (u16)(r >> 16);
}

__device__ __forceinline__ v8f zero8() {
  v8f z = {0.f,0.f,0.f,0.f,0.f,0.f,0.f,0.f};
  return z;
}

// A-matrix 16x32 bf16 fragment (row-major source, lda in elements).
// Lanes 0-15: M=0..15, K = {0..7, 16..23}; lanes 16-31: M=0..15, K = {8..15, 24..31}.
__device__ __forceinline__ v16bf load_frag_a(const u16* __restrict__ base, int lda, int lane) {
  const int m    = lane & 15;
  const int koff = (lane >> 4) << 3;            // 0 or 8
  const u16* p = base + (size_t)m * lda + koff;
  v8u c0 = *(const v8u*)(p);                    // K = koff .. koff+7
  v8u c1 = *(const v8u*)(p + 16);               // K = koff+16 .. koff+23
  v16u r = __builtin_shufflevector(c0, c1, 0,1,2,3,4,5,6,7,8,9,10,11,12,13,14,15);
  return __builtin_bit_cast(v16bf, r);
}

// B-matrix 32x16 bf16 fragment from B^T row-major [n][k] (ldb in elements).
__device__ __forceinline__ v16bf load_frag_b(const u16* __restrict__ baseT, int ldb, int lane) {
  const int n    = lane & 15;
  const int koff = (lane >> 4) << 3;            // 0 or 8
  const u16* p = baseT + (size_t)n * ldb + koff;
  v8u c0 = *(const v8u*)(p);                    // K = koff .. koff+7
  v8u c1 = *(const v8u*)(p + 16);               // K = koff+16 .. koff+23
  v16u r = __builtin_shufflevector(c0, c1, 0,1,2,3,4,5,6,7,8,9,10,11,12,13,14,15);
  return __builtin_bit_cast(v16bf, r);
}

__device__ __forceinline__ v8f wmma_bf16(v16bf a, v16bf b, v8f c) {
  return __builtin_amdgcn_wmma_f32_16x16x32_bf16(false, a, false, b, (short)0, c, false, false);
}

__device__ __forceinline__ void async_join() {
#if defined(HAVE_ASYNC_COPY)
#if __has_builtin(__builtin_amdgcn_s_wait_asynccnt)
  __builtin_amdgcn_s_wait_asynccnt(0);
#else
  asm volatile("s_wait_asynccnt 0x0" ::: "memory");
#endif
#endif
}

__device__ __forceinline__ float wred_max(float v){ for (int o=16;o>0;o>>=1) v = fmaxf(v, __shfl_xor(v,o,32)); return v; }
__device__ __forceinline__ float wred_min(float v){ for (int o=16;o>0;o>>=1) v = fminf(v, __shfl_xor(v,o,32)); return v; }
__device__ __forceinline__ float wred_sum(float v){ for (int o=16;o>0;o>>=1) v +=      __shfl_xor(v,o,32); return v; }
__device__ __forceinline__ int   wred_isum(int v) { for (int o=16;o>0;o>>=1) v +=      __shfl_xor(v,o,32); return v; }

// ---------------- weight transpose + fp32->bf16 ----------------
// W [K][N] fp32 -> Wt [N][K] bf16
__global__ __launch_bounds__(256) void cvt_t(const float* __restrict__ W, u16* __restrict__ Wt,
                                             int K, int N) {
  size_t idx = (size_t)blockIdx.x * 256 + threadIdx.x;
  size_t total = (size_t)K * N;
  if (idx >= total) return;
  int n = (int)(idx / K);
  int k = (int)(idx % K);
  Wt[idx] = f2bf(W[(size_t)k * N + n]);
}

// ---------------- LayerNorm -> bf16 ----------------
__global__ __launch_bounds__(256) void ln_bf16(const float* __restrict__ x,
                                               const float* __restrict__ g,
                                               const float* __restrict__ be,
                                               u16* __restrict__ out) {
  __shared__ float r1[256], r2[256];
  const int row = blockIdx.x;
  const int tid = threadIdx.x;
  const float* xr = x + (size_t)row * EMBED;
  float s = 0.f, s2 = 0.f;
  for (int i = tid; i < EMBED; i += 256) { float v = xr[i]; s += v; s2 += v*v; }
  r1[tid] = s; r2[tid] = s2;
  __syncthreads();
  for (int o = 128; o > 0; o >>= 1) {
    if (tid < o) { r1[tid] += r1[tid+o]; r2[tid] += r2[tid+o]; }
    __syncthreads();
  }
  const float mean = r1[0] * (1.0f/EMBED);
  const float var  = r2[0] * (1.0f/EMBED) - mean*mean;
  const float rstd = rsqrtf(var + 1e-5f);
  u16* orow = out + (size_t)row * EMBED;
  for (int i = tid; i < EMBED; i += 256)
    orow[i] = f2bf((xr[i] - mean) * rstd * g[i] + be[i]);
}

// ---------------- generic bf16 WMMA GEMM with fused epilogues ----------------
// C[M x Nout] = A[M x K] * Bt[Nout x K]^T   (+ bias, + epilogue by mode)
// grid = (M/16, Nout/512), block = 256 (8 waves); each wave does a 16x64 strip.
// The 16xK A panel is staged through LDS with double-buffered ASYNC global->LDS
// copies (ASYNCcnt / s_wait_asynccnt) and consumed via ds_load_b128 fragments,
// so all 8 waves share one copy of A.
#define MODE_Q   0
#define MODE_K   1
#define MODE_VT  2
#define MODE_WO  3
#define MODE_FF1 4
#define MODE_FF2 5

__global__ __launch_bounds__(256) void gemm_bf16_wmma(
    const u16* __restrict__ A, const u16* __restrict__ Bt,
    const float* __restrict__ bias,
    int M, int Nout, int K, int mode,
    float* __restrict__ outf, const float* __restrict__ resid,
    u16* __restrict__ outb) {
  __shared__ alignas(16) u16 At[2][16 * 32];    // double-buffered 16x32 bf16 A chunk

  const int tid  = threadIdx.x;
  const int lane = tid & 31;
  const int wave = tid >> 5;
  const int mbase   = blockIdx.x * 16;
  const int colbase = blockIdx.y * 512 + wave * 64;
  if (mbase >= M) return;                        // block-uniform guard

  const u16* Arow = A + (size_t)mbase * K;

  // stage A chunk [16 x 32] at column kb into buffer buf (1KB, 4B per thread)
  auto stage = [&](int buf, int kb) {
    const int row = tid >> 4;                    // 0..15
    const int col = (tid & 15) << 1;             // 0,2,..,30
    const u16* src = Arow + (size_t)row * K + kb + col;
    u16* dst = &At[buf][row * 32 + col];
#if defined(HAVE_ASYNC_COPY)
    __builtin_amdgcn_global_load_async_to_lds_b32(
        (__attribute__((address_space(1))) int*)src,
        (__attribute__((address_space(3))) int*)dst, 0, 0);
#else
    *(unsigned*)dst = *(const unsigned*)src;
#endif
  };

  v8f acc[4] = { zero8(), zero8(), zero8(), zero8() };

  stage(0, 0);
  async_join();
  __syncthreads();

  for (int kb = 0; kb < K; kb += 32) {
    const int cur = (kb >> 5) & 1;
    if (kb + 32 < K) stage(cur ^ 1, kb + 32);    // overlap copy with WMMA work
    __builtin_prefetch(Bt + (size_t)colbase * K + kb + 32, 0, 1);  // global_prefetch_b8

    v16bf a = load_frag_a(&At[cur][0], 32, lane);   // ds_load_b128 x2
#pragma unroll
    for (int t = 0; t < 4; ++t) {
      v16bf b = load_frag_b(Bt + (size_t)(colbase + t*16) * K + kb, K, lane);
      acc[t] = wmma_bf16(a, b, acc[t]);
    }
    async_join();                                // next buffer landed
    __syncthreads();
  }

  const int nloc = lane & 15;
  const int mo   = (lane >> 4) * 8;
#pragma unroll
  for (int t = 0; t < 4; ++t) {
#pragma unroll
    for (int r = 0; r < 8; ++r) {
      const int m = mbase + mo + r;            // token index
      const int c = colbase + t*16 + nloc;     // output column
      float v = acc[t][r] + bias[c];
      if (mode == MODE_Q || mode == MODE_K) {
        const int bb = m >> 11, n = m & (SEQ-1);
        const int h = c >> 6,  d = c & (HDIM-1);
        outb[((((size_t)bb*NHEAD + h)*SEQ + n) << 6) + d] = f2bf(v);
      } else if (mode == MODE_VT) {
        const int bb = m >> 11, n = m & (SEQ-1);
        const int h = c >> 6,  d = c & (HDIM-1);
        outb[(((size_t)bb*NHEAD + h)*HDIM + d)*SEQ + n] = f2bf(v);
      } else if (mode == MODE_WO) {
        const size_t idx = (size_t)m * EMBED + c;
        outf[idx] = v + resid[idx];
      } else if (mode == MODE_FF1) {
        const float ge = 0.5f * v * (1.0f + erff(v * 0.70710678118654752f));
        outb[(size_t)m * DFF + c] = f2bf(ge);
      } else { // MODE_FF2
        const size_t idx = (size_t)m * EMBED + c;
        outf[idx] = outf[idx] + v;             // x1 + ff
      }
    }
  }
}

// ---------------- top-k sparse attention ----------------
// grid = (SEQ/16, NHEAD, BATCH), block = 256 (8 waves).
// Dynamic LDS: 16*2048 fp32 scores (128 KB of the 320 KB WGP LDS) + row scales
// + 2 KB Q-tile staged by the Tensor Data Mover (tensor_load_to_lds, TENSORcnt).
#define ATTN_LDS_BYTES ((size_t)16 * SEQ * 4 + 64 + 16 * HDIM * 2)

__global__ __launch_bounds__(256) void attn_topk(const u16* __restrict__ Qb,
                                                 const u16* __restrict__ Kb,
                                                 const u16* __restrict__ Vtb,
                                                 u16* __restrict__ ctx) {
  extern __shared__ float smem[];
  float* S      = smem;                 // [16][2048]
  float* rowinv = smem + 16*SEQ;        // [16]
  u16*   Qs     = (u16*)(rowinv + 16);  // [16][64] TDM-staged Q tile

  const int lane = threadIdx.x & 31;
  const int wave = threadIdx.x >> 5;
  const int qt = blockIdx.x, h = blockIdx.y, b = blockIdx.z;

  const size_t head = (size_t)(b*NHEAD + h);
  const u16* Qh = Qb  + head * SEQ * HDIM + (size_t)qt * 16 * HDIM;
  const u16* Kh = Kb  + head * SEQ * HDIM;
  const u16* Vh = Vtb + head * HDIM * SEQ;   // transposed: [d][n]

  // ---- Stage the 16x64 Q tile into LDS via the Tensor Data Mover ----
#if defined(HAVE_TDM)
  if (wave == 0) {
    const unsigned long long ga = (unsigned long long)(uintptr_t)Qh;
    const unsigned lds_off = (unsigned)(unsigned long long)(uintptr_t)(void*)Qs; // low 32b = LDS offset
    v4ui g0;
    g0.x = 1u;                                            // count=1, user descriptor
    g0.y = lds_off;                                       // D#.lds_addr
    g0.z = (unsigned)(ga & 0xFFFFFFFFu);                  // D#.global_addr[31:0]
    g0.w = (unsigned)((ga >> 32) & 0x01FFFFFFu)           // D#.global_addr[56:32]
         | (2u << 30);                                    // type = 2 ("image")
    v8i g1;
    g1[0] = (int)(1u << 16);        // workgroup_mask=0, data_size=1 (2 bytes)
    g1[1] = (int)((unsigned)HDIM << 16);   // tensor_dim0[15:0] = 64
    g1[2] = (int)((unsigned)SEQ  << 16);   // tensor_dim0 hi=0, tensor_dim1[15:0] = 2048
    g1[3] = (int)((unsigned)HDIM << 16);   // tensor_dim1 hi=0, tile_dim0 = 64
    g1[4] = 16;                     // tile_dim1 = 16, tile_dim2 = 0
    g1[5] = HDIM;                   // tensor_dim0_stride[31:0] = 64
    g1[6] = 0;                      // stride0 hi, stride1 lo
    g1[7] = 0;                      // stride1 hi
    v4i z4 = {0,0,0,0};
#if __clang_major__ >= 23
    v8i z8 = {0,0,0,0,0,0,0,0};
    __builtin_amdgcn_tensor_load_to_lds(g0, g1, z4, z4, z8, 0);
#else
    __builtin_amdgcn_tensor_load_to_lds(g0, g1, z4, z4, 0);
#endif
    __builtin_amdgcn_s_wait_tensorcnt(0);
  }
  __syncthreads();
  const v16bf a0 = load_frag_a(Qs +  0, HDIM, lane);   // ds_load_b128 from TDM tile
  const v16bf a1 = load_frag_a(Qs + 32, HDIM, lane);
#else
  const v16bf a0 = load_frag_a(Qh +  0, HDIM, lane);
  const v16bf a1 = load_frag_a(Qh + 32, HDIM, lane);
#endif

  // Phase 1: S = (Q K^T) * 1/sqrt(64)
  for (int ct = wave; ct < SEQ/16; ct += 8) {
    v8f acc = zero8();
    v16bf b0 = load_frag_b(Kh + (size_t)ct*16*HDIM +  0, HDIM, lane);
    acc = wmma_bf16(a0, b0, acc);
    v16bf b1 = load_frag_b(Kh + (size_t)ct*16*HDIM + 32, HDIM, lane);
    acc = wmma_bf16(a1, b1, acc);
    const int n0 = ct*16 + (lane & 15);
    const int mo = (lane >> 4) * 8;
#pragma unroll
    for (int r = 0; r < 8; ++r) S[(mo + r)*SEQ + n0] = acc[r] * 0.125f;
  }
  __syncthreads();

  // Phase 2: per-row top-64 threshold (binary search on value) + masked exp
  for (int row = wave; row < 16; row += 8) {
    float* Sr = S + (size_t)row * SEQ;
    float mx = -3.4e38f, mn = 3.4e38f;
    for (int i = lane; i < SEQ; i += 32) { float s = Sr[i]; mx = fmaxf(mx, s); mn = fminf(mn, s); }
    mx = wred_max(mx); mn = wred_min(mn);
    float lo = mn, hi = mx;
    for (int it = 0; it < 24; ++it) {
      const float mid = 0.5f * (lo + hi);
      int cnt = 0;
      for (int i = lane; i < SEQ; i += 32) cnt += (Sr[i] >= mid) ? 1 : 0;
      cnt = wred_isum(cnt);
      if (cnt >= TOPK) lo = mid; else hi = mid;
    }
    const float thr = lo;
    float sum = 0.f;
    for (int i = lane; i < SEQ; i += 32) {
      const float s = Sr[i];
      const float p = (s >= thr) ? __expf(s - mx) : 0.f;
      Sr[i] = p; sum += p;
    }
    sum = wred_sum(sum);
    if (lane == 0) rowinv[row] = 1.0f / sum;
  }
  __syncthreads();

  // Phase 3: out = P V (waves 0..3, one 16x16 d-tile each, full K=2048)
  if (wave < 4) {
    v8f acc = zero8();
    const int m    = lane & 15;
    const int koff = (lane >> 4) << 3;
    const float inv = rowinv[m];
    for (int kb = 0; kb < SEQ; kb += 32) {
      const float* p = S + (size_t)m * SEQ + kb + koff;
      v16u ev;
#pragma unroll
      for (int i = 0; i < 8; ++i) ev[i]   = f2bf(p[i]      * inv);
#pragma unroll
      for (int i = 0; i < 8; ++i) ev[8+i] = f2bf(p[16 + i] * inv);
      v16bf a = __builtin_bit_cast(v16bf, ev);
      v16bf bf = load_frag_b(Vh + (size_t)(wave*16)*SEQ + kb, SEQ, lane);
      acc = wmma_bf16(a, bf, acc);
    }
    const int d  = wave*16 + (lane & 15);
    const int mo = (lane >> 4) * 8;
#pragma unroll
    for (int r = 0; r < 8; ++r) {
      const int nq = qt*16 + mo + r;
      ctx[((((size_t)b*SEQ + nq)*NHEAD + h) << 6) + d] = f2bf(acc[r]);
    }
  }
}

// ---------------- host orchestration ----------------
extern "C" void kernel_launch(void* const* d_in, const int* in_sizes, int n_in,
                              void* d_out, int out_size, void* d_ws, size_t ws_size,
                              hipStream_t stream) {
  const float* x   = (const float*)d_in[0];
  const float* Wq  = (const float*)d_in[1];
  const float* bq  = (const float*)d_in[2];
  const float* Wk  = (const float*)d_in[3];
  const float* bk  = (const float*)d_in[4];
  const float* Wv  = (const float*)d_in[5];
  const float* bv  = (const float*)d_in[6];
  const float* Wo  = (const float*)d_in[7];
  const float* bo  = (const float*)d_in[8];
  const float* g1  = (const float*)d_in[9];
  const float* be1 = (const float*)d_in[10];
  const float* g2  = (const float*)d_in[11];
  const float* be2 = (const float*)d_in[12];
  const float* W1  = (const float*)d_in[13];
  const float* b1  = (const float*)d_in[14];
  const float* W2  = (const float*)d_in[15];
  const float* b2  = (const float*)d_in[16];
  float* out = (float*)d_out;

  // workspace carve (256B aligned)
  char* w = (char*)d_ws;
  auto carve = [&](size_t bytes) -> void* {
    void* p = (void*)w;
    w += (bytes + 255) & ~(size_t)255;
    return p;
  };
  u16* hb   = (u16*)carve((size_t)TOK * EMBED * 2);
  u16* WqT  = (u16*)carve((size_t)EMBED * EMBED * 2);
  u16* WkT  = (u16*)carve((size_t)EMBED * EMBED * 2);
  u16* WvT  = (u16*)carve((size_t)EMBED * EMBED * 2);
  u16* WoT  = (u16*)carve((size_t)EMBED * EMBED * 2);
  u16* W1T  = (u16*)carve((size_t)DFF * EMBED * 2);
  u16* W2T  = (u16*)carve((size_t)EMBED * DFF * 2);
  u16* Qb   = (u16*)carve((size_t)TOK * EMBED * 2);
  u16* Kbuf = (u16*)carve((size_t)TOK * EMBED * 2);
  u16* Vt   = (u16*)carve((size_t)TOK * EMBED * 2);
  u16* ctx  = (u16*)carve((size_t)TOK * EMBED * 2);
  u16* h2   = (u16*)carve((size_t)TOK * EMBED * 2);
  u16* ffa  = (u16*)carve((size_t)TOK * DFF * 2);

  // 1) weight transposes -> bf16
  {
    const int n_ee  = EMBED*EMBED;
    cvt_t<<<(n_ee+255)/256, 256, 0, stream>>>(Wq, WqT, EMBED, EMBED);
    cvt_t<<<(n_ee+255)/256, 256, 0, stream>>>(Wk, WkT, EMBED, EMBED);
    cvt_t<<<(n_ee+255)/256, 256, 0, stream>>>(Wv, WvT, EMBED, EMBED);
    cvt_t<<<(n_ee+255)/256, 256, 0, stream>>>(Wo, WoT, EMBED, EMBED);
    const int n_ed = EMBED*DFF;
    cvt_t<<<(n_ed+255)/256, 256, 0, stream>>>(W1, W1T, EMBED, DFF);   // -> [DFF][EMBED]
    cvt_t<<<(n_ed+255)/256, 256, 0, stream>>>(W2, W2T, DFF, EMBED);   // -> [EMBED][DFF]
  }

  // 2) LN1
  ln_bf16<<<TOK, 256, 0, stream>>>(x, g1, be1, hb);

  // 3) QKV projections (Q/K head-major, V transposed per head)
  {
    dim3 grid(TOK/16, EMBED/512);
    gemm_bf16_wmma<<<grid, 256, 0, stream>>>(hb, WqT, bq, TOK, EMBED, EMBED, MODE_Q,  nullptr, nullptr, Qb);
    gemm_bf16_wmma<<<grid, 256, 0, stream>>>(hb, WkT, bk, TOK, EMBED, EMBED, MODE_K,  nullptr, nullptr, Kbuf);
    gemm_bf16_wmma<<<grid, 256, 0, stream>>>(hb, WvT, bv, TOK, EMBED, EMBED, MODE_VT, nullptr, nullptr, Vt);
  }

  // 4) top-k sparse attention
  {
    dim3 grid(SEQ/16, NHEAD, BATCH);
    attn_topk<<<grid, 256, ATTN_LDS_BYTES, stream>>>(Qb, Kbuf, Vt, ctx);
  }

  // 5) output projection + residual -> x1 (fp32 in d_out)
  {
    dim3 grid(TOK/16, EMBED/512);
    gemm_bf16_wmma<<<grid, 256, 0, stream>>>(ctx, WoT, bo, TOK, EMBED, EMBED, MODE_WO, out, x, nullptr);
  }

  // 6) LN2 on x1
  ln_bf16<<<TOK, 256, 0, stream>>>(out, g2, be2, h2);

  // 7) FFN: gelu(h2 W1 + b1) W2 + b2, accumulate into d_out
  {
    dim3 g1d(TOK/16, DFF/512);
    gemm_bf16_wmma<<<g1d, 256, 0, stream>>>(h2, W1T, b1, TOK, DFF, EMBED, MODE_FF1, nullptr, nullptr, ffa);
    dim3 g2d(TOK/16, EMBED/512);
    gemm_bf16_wmma<<<g2d, 256, 0, stream>>>(ffa, W2T, b2, TOK, EMBED, DFF, MODE_FF2, out, nullptr, nullptr);
  }
}
